// GatedEquivariantConv_32959579030386
// MI455X (gfx1250) — matrix operand order repaired and checked by reading, compiled, not run on hardware
//
#include <hip/hip_runtime.h>

typedef __attribute__((ext_vector_type(16))) __bf16 v16bf;
typedef __attribute__((ext_vector_type(8)))  float  v8f;

#define NNODE 32768
#define NEDGE 262144
#define NSC   128
#define NVC   64
#define DDIM  320
#define SQ3   1.7320508075688772f

__device__ __forceinline__ float frcp(float x){ return __builtin_amdgcn_rcpf(x); }
__device__ __forceinline__ float sigmf(float x){ return frcp(1.0f + __expf(-x)); }
__device__ __forceinline__ float siluf(float x){ return x * sigmf(x); }

__device__ __forceinline__ v8f wmma_bf(v16bf a, v16bf b, v8f c){
  return __builtin_amdgcn_wmma_f32_16x16x32_bf16(false, a, false, b, (short)0, c, false, false);
}

// A fragment (16x32 bf16): p -> (row0, k0) of the tile, stride = row pitch.
// Layout per ISA: lanes 0-15 rows 0-15, VGPR0-3 K=kb..kb+7 (kb=0), VGPR4-7 K=16+kb..;
// lanes 16-31 kb=8.  Contiguous per lane -> compiler emits ds_load_b128 x2.
__device__ __forceinline__ v16bf lds_load_a(const __bf16* p, int stride){
  const int lane = threadIdx.x & 31;
  const int row  = lane & 15;
  const int kb   = (lane & 16) ? 8 : 0;
  v16bf a;
#pragma unroll
  for (int j = 0; j < 16; ++j){
    int k = ((j & 8) ? 16 : 0) + kb + (j & 7);
    a[j] = p[row * stride + k];
  }
  return a;
}

// B fragment from repacked weights: 512 halfs/fragment, lane-contiguous 16 halfs.
__device__ __forceinline__ v16bf gbl_load_b(const __bf16* w, int frag){
  const int lane = threadIdx.x & 31;
  return *(const v16bf*)(w + ((size_t)frag * 32 + lane) * 16);
}

// ---------------------------------------------------------------- repack ----
__global__ void repack_kernel(const float* __restrict__ src, __bf16* __restrict__ dst,
                              int K, int Nsrc, int Npad){
  const int KT = K >> 5;
  const int total = KT * (Npad >> 4) * 512;
  for (int idx = blockIdx.x * blockDim.x + threadIdx.x; idx < total;
       idx += gridDim.x * blockDim.x){
    int frag = idx >> 9, rem = idx & 511;
    int lane = rem >> 4, j = rem & 15;
    int nt = frag / KT, kt = frag % KT;
    int n = nt * 16 + (lane & 15);
    int k = kt * 32 + ((j & 8) ? 16 : 0) + ((lane & 16) ? 8 : 0) + (j & 7);
    float v = (n < Nsrc) ? src[(size_t)k * Nsrc + n] : 0.0f;
    dst[idx] = (__bf16)v;
  }
}

__global__ void zero_f32(float* p, size_t n){
  size_t i = (size_t)blockIdx.x * blockDim.x + threadIdx.x;
  size_t st = (size_t)gridDim.x * blockDim.x;
  for (; i < n; i += st) p[i] = 0.0f;
}

// ------------------------------------------------------------ edge kernel ---
__global__ __launch_bounds__(32)
void edge_kernel(const float* __restrict__ h,
                 const float* __restrict__ edge_vec,
                 const float* __restrict__ edge_scalars,
                 const float* __restrict__ edge_dist,
                 const int*   __restrict__ src_idx,
                 const int*   __restrict__ dst_idx,
                 const int*   __restrict__ edge_type,
                 const float* __restrict__ log_sigma,
                 const float* __restrict__ rt_b,   const float* __restrict__ rin_b,
                 const float* __restrict__ rout_b,
                 const float* __restrict__ act_b1, const float* __restrict__ act_b2,
                 const float* __restrict__ gate_b1,const float* __restrict__ gate_b2,
                 const __bf16* __restrict__ rtW,  const __bf16* __restrict__ rinW,
                 const __bf16* __restrict__ routW,
                 const __bf16* __restrict__ tp00W,const __bf16* __restrict__ tp01W,
                 const __bf16* __restrict__ tp10W,const __bf16* __restrict__ tp11W,
                 const __bf16* __restrict__ act1W,const __bf16* __restrict__ act2W,
                 const __bf16* __restrict__ g1W,  const __bf16* __restrict__ g2W,
                 float* __restrict__ sum_s,  float* __restrict__ sum_sg,
                 float* __restrict__ sum_v,  float* __restrict__ sum_vg)
{
  __shared__ __bf16 s_es[16][64];
  __shared__ __bf16 s_tr[16][128];
  __shared__ __bf16 s_xs[16][128];
  __shared__ __bf16 s_xv[3][16][64];
  __shared__ __bf16 s_dot[16][64];
  __shared__ __bf16 s_so[16][192];
  __shared__ __bf16 s_mv[3][16][64];
  __shared__ __bf16 s_xs01[16][64];
  __shared__ __bf16 s_vn[16][64];
  __shared__ __bf16 s_t1[16][64];
  __shared__ __bf16 s_t2[16][64];
  __shared__ __bf16 s_gg[16][64];
  __shared__ __bf16 s_vg[16][64];
  __shared__ float  s_y1[16][3];
  __shared__ float  s_sg[16];
  __shared__ int    s_src[16];
  __shared__ int    s_dst[16];

  const int lane = threadIdx.x & 31;
  const int e0   = blockIdx.x * 16;
  const int ln   = lane & 15;
  const int mh   = (lane & 16) ? 8 : 0;

  // --- phase 0: per-edge scalars + edge_scalars tile -------------------------
  if (lane < 16){
    int e = e0 + lane;
    s_src[lane] = src_idx[e];
    s_dst[lane] = dst_idx[e];
    float vx = edge_vec[e*3+0], vy = edge_vec[e*3+1], vz = edge_vec[e*3+2];
    float inv = SQ3 * __builtin_amdgcn_rsqf(vx*vx + vy*vy + vz*vz + 1e-12f);
    s_y1[lane][0] = vx*inv; s_y1[lane][1] = vy*inv; s_y1[lane][2] = vz*inv;
    float inv_sigma = __expf(-log_sigma[edge_type[e]]);       // 1/sigma
    s_sg[lane] = __expf(-edge_dist[e] * inv_sigma);           // decay
  }
#pragma unroll
  for (int q = 0; q < 16; ++q){
    int idx = lane * 16 + q;                       // 0..511
    s_es[idx >> 6][idx & 63] = (__bf16)edge_scalars[(size_t)e0 * 64 + idx];
  }
  __syncthreads();

  // --- phase 1: trunk = silu(es@rt_w+b), t2 = silu(es@gate_w1+b) -------------
#pragma unroll 1
  for (int nt = 0; nt < 8; ++nt){
    v8f acc = {};
#pragma unroll
    for (int kt = 0; kt < 2; ++kt)
      acc = wmma_bf(lds_load_a(&s_es[0][kt*32], 64), gbl_load_b(rtW, nt*2+kt), acc);
    int n = nt*16 + ln; float bias = rt_b[n];
#pragma unroll
    for (int r = 0; r < 8; ++r) s_tr[mh+r][n] = (__bf16)siluf(acc[r] + bias);
  }
#pragma unroll 1
  for (int nt = 0; nt < 4; ++nt){
    v8f acc = {};
#pragma unroll
    for (int kt = 0; kt < 2; ++kt)
      acc = wmma_bf(lds_load_a(&s_es[0][kt*32], 64), gbl_load_b(g1W, nt*2+kt), acc);
    int n = nt*16 + ln; float bias = gate_b1[n];
#pragma unroll
    for (int r = 0; r < 8; ++r) s_t2[mh+r][n] = (__bf16)siluf(acc[r] + bias);
  }
  __syncthreads();

  // --- phase 2: scale_in = 1+expand(trunk@rin_w+b); hs = h[src]*scale_in -----
#pragma unroll 1
  for (int nt = 0; nt < 12; ++nt){
    v8f acc = {};
#pragma unroll
    for (int kt = 0; kt < 4; ++kt)
      acc = wmma_bf(lds_load_a(&s_tr[0][kt*32], 128), gbl_load_b(rinW, nt*4+kt), acc);
    int n = nt*16 + ln; float bias = rin_b[n];
#pragma unroll
    for (int r = 0; r < 8; ++r){
      int m = mh + r;
      float sc = 1.0f + acc[r] + bias;
      const float* hb = h + (size_t)s_src[m] * DDIM;
      if (n < NSC){
        s_xs[m][n] = (__bf16)(hb[n] * sc);
      } else {
        int w = n - NSC;
#pragma unroll
        for (int i = 0; i < 3; ++i)
          s_xv[i][m][w] = (__bf16)(hb[NSC + 3*w + i] * sc);
      }
    }
  }
  __syncthreads();

  // --- phase 3: dot, scale_out, go/gates -------------------------------------
#pragma unroll
  for (int q = 0; q < 16; ++q){
    int idx = lane*16 + q; int m = idx >> 6, u = idx & 63;
    float d = ((float)s_xv[0][m][u]*s_y1[m][0] + (float)s_xv[1][m][u]*s_y1[m][1]
             + (float)s_xv[2][m][u]*s_y1[m][2]) * (1.0f/SQ3);
    s_dot[m][u] = (__bf16)d;
  }
#pragma unroll 1
  for (int nt = 0; nt < 12; ++nt){
    v8f acc = {};
#pragma unroll
    for (int kt = 0; kt < 4; ++kt)
      acc = wmma_bf(lds_load_a(&s_tr[0][kt*32], 128), gbl_load_b(routW, nt*4+kt), acc);
    int n = nt*16 + ln; float bias = rout_b[n];
#pragma unroll
    for (int r = 0; r < 8; ++r) s_so[mh+r][n] = (__bf16)(1.0f + acc[r] + bias);
  }
#pragma unroll 1
  for (int nt = 0; nt < 5; ++nt){                      // go, N padded 65->80
    v8f acc = {};
#pragma unroll
    for (int kt = 0; kt < 2; ++kt)
      acc = wmma_bf(lds_load_a(&s_t2[0][kt*32], 64), gbl_load_b(g2W, nt*2+kt), acc);
    int n = nt*16 + ln; float bias = (n < 65) ? gate_b2[n] : 0.0f;
#pragma unroll
    for (int r = 0; r < 8; ++r){
      int m = mh + r;
      float val = acc[r] + bias;
      if (n == 0)      s_sg[m] = s_sg[m] * sigmf(val);
      else if (n < 65) s_vg[m][n-1] = (__bf16)sigmf(val);
    }
  }
  __syncthreads();

  // --- phase 4: xs01 = xs@tp01; msg_s scatter --------------------------------
#pragma unroll 1
  for (int nt = 0; nt < 4; ++nt){
    v8f acc = {};
#pragma unroll
    for (int kt = 0; kt < 4; ++kt)
      acc = wmma_bf(lds_load_a(&s_xs[0][kt*32], 128), gbl_load_b(tp01W, nt*4+kt), acc);
    int n = nt*16 + ln;
#pragma unroll
    for (int r = 0; r < 8; ++r) s_xs01[mh+r][n] = (__bf16)acc[r];
  }
#pragma unroll 1
  for (int nt = 0; nt < 8; ++nt){
    v8f acc = {};
#pragma unroll
    for (int kt = 0; kt < 4; ++kt)
      acc = wmma_bf(lds_load_a(&s_xs[0][kt*32], 128), gbl_load_b(tp00W, nt*4+kt), acc);
#pragma unroll
    for (int kt = 0; kt < 2; ++kt)
      acc = wmma_bf(lds_load_a(&s_dot[0][kt*32], 64), gbl_load_b(tp11W, nt*2+kt), acc);
    int n = nt*16 + ln;
#pragma unroll
    for (int r = 0; r < 8; ++r){
      int m = mh + r;
      float val = siluf(acc[r] * (float)s_so[m][n]) * s_sg[m];
      atomicAdd(&sum_s[(size_t)s_dst[m] * NSC + n], val);
    }
  }
  if (lane < 16) atomicAdd(&sum_sg[s_dst[lane]], s_sg[lane]);
  __syncthreads();

  // --- phase 5: msg_v --------------------------------------------------------
#pragma unroll 1
  for (int i = 0; i < 3; ++i){
#pragma unroll 1
    for (int nt = 0; nt < 4; ++nt){
      v8f acc = {};
#pragma unroll
      for (int kt = 0; kt < 2; ++kt)
        acc = wmma_bf(lds_load_a(&s_xv[i][0][kt*32], 64), gbl_load_b(tp10W, nt*2+kt), acc);
      int w = nt*16 + ln;
#pragma unroll
      for (int r = 0; r < 8; ++r){
        int m = mh + r;
        float val = ((float)s_xs01[m][w] * s_y1[m][i] + acc[r]) * (float)s_so[m][NSC + w];
        s_mv[i][m][w] = (__bf16)val;
      }
    }
  }
  __syncthreads();
#pragma unroll
  for (int q = 0; q < 16; ++q){
    int idx = lane*16 + q; int m = idx >> 6, w = idx & 63;
    float a0 = (float)s_mv[0][m][w], a1 = (float)s_mv[1][m][w], a2 = (float)s_mv[2][m][w];
    s_vn[m][w] = (__bf16)__builtin_amdgcn_sqrtf(a0*a0 + a1*a1 + a2*a2 + 1e-12f);
  }
  __syncthreads();
#pragma unroll 1
  for (int nt = 0; nt < 4; ++nt){                      // t1 = silu(vn@act_w1+b1)
    v8f acc = {};
#pragma unroll
    for (int kt = 0; kt < 2; ++kt)
      acc = wmma_bf(lds_load_a(&s_vn[0][kt*32], 64), gbl_load_b(act1W, nt*2+kt), acc);
    int n = nt*16 + ln; float bias = act_b1[n];
#pragma unroll
    for (int r = 0; r < 8; ++r) s_t1[mh+r][n] = (__bf16)siluf(acc[r] + bias);
  }
  __syncthreads();
#pragma unroll 1
  for (int nt = 0; nt < 4; ++nt){                      // g = sigmoid(t1@act_w2+b2); gg=g*vg
    v8f acc = {};
#pragma unroll
    for (int kt = 0; kt < 2; ++kt)
      acc = wmma_bf(lds_load_a(&s_t1[0][kt*32], 64), gbl_load_b(act2W, nt*2+kt), acc);
    int n = nt*16 + ln; float bias = act_b2[n];
#pragma unroll
    for (int r = 0; r < 8; ++r){
      int m = mh + r;
      s_gg[m][n] = (__bf16)(sigmf(acc[r] + bias) * (float)s_vg[m][n]);
    }
  }
  __syncthreads();
  // --- phase 6: vector scatter ----------------------------------------------
#pragma unroll
  for (int q = 0; q < 16; ++q){
    int idx = lane*16 + q; int m = idx >> 6, w = idx & 63;
    int d = s_dst[m];
    float gv = (float)s_gg[m][w];
    size_t base = (size_t)d * (NVC*3) + (size_t)w * 3;
#pragma unroll
    for (int i = 0; i < 3; ++i)
      atomicAdd(&sum_v[base + i], (float)s_mv[i][m][w] * gv);
    atomicAdd(&sum_vg[(size_t)d * NVC + w], (float)s_vg[m][w]);
  }
}

// ------------------------------------------------------------ node kernel ---
__global__ __launch_bounds__(32)
void node_kernel(const float* __restrict__ h,
                 const float* __restrict__ sum_s, const float* __restrict__ sum_sg,
                 const float* __restrict__ sum_v, const float* __restrict__ sum_vg,
                 const __bf16* __restrict__ vrW,  const float* __restrict__ vr_b,
                 const __bf16* __restrict__ swsW, const float* __restrict__ self_bs,
                 const __bf16* __restrict__ swvW,
                 float* __restrict__ out)
{
  __shared__ __bf16 s_hs[16][128];
  __shared__ __bf16 s_hv[3][16][64];
  __shared__ __bf16 s_vnb[16][64];
  __shared__ float  s_agg[3][16][64];
  __shared__ float  s_vnf[16][64];
  __shared__ float  s_rad[16][64];

  const int lane = threadIdx.x & 31;
  const int n0   = blockIdx.x * 16;
  const int ln   = lane & 15;
  const int mh   = (lane & 16) ? 8 : 0;

#pragma unroll
  for (int q = 0; q < 64; ++q){
    int idx = lane*64 + q; int m = idx >> 7, c = idx & 127;
    s_hs[m][c] = (__bf16)h[(size_t)(n0+m)*DDIM + c];
  }
#pragma unroll
  for (int q = 0; q < 96; ++q){
    int idx = lane*96 + q; int m = idx / 192, rem = idx % 192;
    s_hv[rem % 3][m][rem / 3] = (__bf16)h[(size_t)(n0+m)*DDIM + NSC + rem];
  }
#pragma unroll
  for (int q = 0; q < 96; ++q){
    int idx = lane*96 + q; int m = idx / 192, rem = idx % 192;
    int w = rem / 3, i = rem % 3;
    int node = n0 + m;
    s_agg[i][m][w] = sum_v[(size_t)node*192 + rem] *
                     frcp(sum_vg[(size_t)node*64 + w] + 1e-6f);
  }
  __syncthreads();
#pragma unroll
  for (int q = 0; q < 16; ++q){
    int idx = lane*16 + q; int m = idx >> 6, w = idx & 63;
    float a0 = s_agg[0][m][w], a1 = s_agg[1][m][w], a2 = s_agg[2][m][w];
    float vn = __builtin_amdgcn_sqrtf(a0*a0 + a1*a1 + a2*a2 + 1e-12f);
    s_vnf[m][w] = vn;
    s_vnb[m][w] = (__bf16)vn;
  }
  __syncthreads();
#pragma unroll 1
  for (int nt = 0; nt < 4; ++nt){                      // rad = vn@vr_w + vr_b
    v8f acc = {};
#pragma unroll
    for (int kt = 0; kt < 2; ++kt)
      acc = wmma_bf(lds_load_a(&s_vnb[0][kt*32], 64), gbl_load_b(vrW, nt*2+kt), acc);
    int n = nt*16 + ln; float bias = vr_b[n];
#pragma unroll
    for (int r = 0; r < 8; ++r) s_rad[mh+r][n] = acc[r] + bias;
  }
  __syncthreads();
#pragma unroll 1
  for (int nt = 0; nt < 8; ++nt){                      // out_s
    v8f acc = {};
#pragma unroll
    for (int kt = 0; kt < 4; ++kt)
      acc = wmma_bf(lds_load_a(&s_hs[0][kt*32], 128), gbl_load_b(swsW, nt*4+kt), acc);
    int n = nt*16 + ln; float bias = self_bs[n];
#pragma unroll
    for (int r = 0; r < 8; ++r){
      int m = mh + r, node = n0 + m;
      float segs = sum_s[(size_t)node*NSC + n] * frcp(sum_sg[node] + 1e-6f);
      out[(size_t)node*DDIM + n] = acc[r] + bias + segs;
    }
  }
#pragma unroll 1
  for (int i = 0; i < 3; ++i){                         // out_v
#pragma unroll 1
    for (int nt = 0; nt < 4; ++nt){
      v8f acc = {};
#pragma unroll
      for (int kt = 0; kt < 2; ++kt)
        acc = wmma_bf(lds_load_a(&s_hv[i][0][kt*32], 64), gbl_load_b(swvW, nt*2+kt), acc);
      int w = nt*16 + ln;
#pragma unroll
      for (int r = 0; r < 8; ++r){
        int m = mh + r, node = n0 + m;
        float val = s_agg[i][m][w] * frcp(s_vnf[m][w] + 1e-6f) * s_rad[m][w] + acc[r];
        out[(size_t)node*DDIM + NSC + (size_t)w*3 + i] = val;
      }
    }
  }
}

// ---------------------------------------------------------------- launch ----
extern "C" void kernel_launch(void* const* d_in, const int* in_sizes, int n_in,
                              void* d_out, int out_size, void* d_ws, size_t ws_size,
                              hipStream_t stream){
  const float* h            = (const float*)d_in[0];
  const float* edge_vec     = (const float*)d_in[1];
  const float* edge_scalars = (const float*)d_in[2];
  const float* edge_dist    = (const float*)d_in[3];
  const int*   src_idx      = (const int*)  d_in[4];
  const int*   dst_idx      = (const int*)  d_in[5];
  const int*   edge_type    = (const int*)  d_in[6];
  const float* rt_w    = (const float*)d_in[8];
  const float* rt_b    = (const float*)d_in[9];
  const float* rin_w   = (const float*)d_in[10];
  const float* rin_b   = (const float*)d_in[11];
  const float* rout_w  = (const float*)d_in[12];
  const float* rout_b  = (const float*)d_in[13];
  const float* tp_w00  = (const float*)d_in[14];
  const float* tp_w01  = (const float*)d_in[15];
  const float* tp_w10  = (const float*)d_in[16];
  const float* tp_w11  = (const float*)d_in[17];
  const float* act_w1  = (const float*)d_in[18];
  const float* act_b1  = (const float*)d_in[19];
  const float* act_w2  = (const float*)d_in[20];
  const float* act_b2  = (const float*)d_in[21];
  const float* gate_w1 = (const float*)d_in[22];
  const float* gate_b1 = (const float*)d_in[23];
  const float* gate_w2 = (const float*)d_in[24];
  const float* gate_b2 = (const float*)d_in[25];
  const float* log_sigma = (const float*)d_in[26];
  const float* vr_w    = (const float*)d_in[27];
  const float* vr_b    = (const float*)d_in[28];
  const float* self_ws = (const float*)d_in[29];
  const float* self_bs = (const float*)d_in[30];
  const float* self_wv = (const float*)d_in[31];

  const int E = in_sizes[3];            // edge_dist count
  const int N = in_sizes[0] / DDIM;     // node count
  float* out = (float*)d_out;

  char* ws = (char*)d_ws;
  auto alloc = [&](size_t bytes)->char*{
    char* p = ws; ws += (bytes + 255) & ~(size_t)255; return p; };

  __bf16* rtW   = (__bf16*)alloc(64  * 128 * 2);
  __bf16* rinW  = (__bf16*)alloc(128 * 192 * 2);
  __bf16* routW = (__bf16*)alloc(128 * 192 * 2);
  __bf16* tp00W = (__bf16*)alloc(128 * 128 * 2);
  __bf16* tp01W = (__bf16*)alloc(128 * 64  * 2);
  __bf16* tp10W = (__bf16*)alloc(64  * 64  * 2);
  __bf16* tp11W = (__bf16*)alloc(64  * 128 * 2);
  __bf16* act1W = (__bf16*)alloc(64  * 64  * 2);
  __bf16* act2W = (__bf16*)alloc(64  * 64  * 2);
  __bf16* g1W   = (__bf16*)alloc(64  * 64  * 2);
  __bf16* g2W   = (__bf16*)alloc(64  * 80  * 2);   // 65 padded to 80
  __bf16* vrW   = (__bf16*)alloc(64  * 64  * 2);
  __bf16* swsW  = (__bf16*)alloc(128 * 128 * 2);
  __bf16* swvW  = (__bf16*)alloc(64  * 64  * 2);

  float* sum_s  = (float*)alloc((size_t)N * 128 * 4);
  float* sum_sg = (float*)alloc((size_t)N * 4);
  float* sum_v  = (float*)alloc((size_t)N * 192 * 4);
  float* sum_vg = (float*)alloc((size_t)N * 64  * 4);

  auto rp = [&](const float* s, __bf16* dptr, int K, int Nsrc, int Npad){
    int total = (K/32) * (Npad/16) * 512;
    repack_kernel<<<(total + 255) / 256, 256, 0, stream>>>(s, dptr, K, Nsrc, Npad);
  };
  rp(rt_w,   rtW,   64, 128, 128);
  rp(rin_w,  rinW,  128, 192, 192);
  rp(rout_w, routW, 128, 192, 192);
  rp(tp_w00, tp00W, 128, 128, 128);
  rp(tp_w01, tp01W, 128, 64,  64);
  rp(tp_w10, tp10W, 64,  64,  64);
  rp(tp_w11, tp11W, 64,  128, 128);
  rp(act_w1, act1W, 64,  64,  64);
  rp(act_w2, act2W, 64,  64,  64);
  rp(gate_w1,g1W,   64,  64,  64);
  rp(gate_w2,g2W,   64,  65,  80);
  rp(vr_w,   vrW,   64,  64,  64);
  rp(self_ws,swsW,  128, 128, 128);
  rp(self_wv,swvW,  64,  64,  64);

  size_t acc_floats = (size_t)N * (128 + 1 + 192 + 64);
  zero_f32<<<2048, 256, 0, stream>>>(sum_s, acc_floats);

  edge_kernel<<<E / 16, 32, 0, stream>>>(
      h, edge_vec, edge_scalars, edge_dist, src_idx, dst_idx, edge_type, log_sigma,
      rt_b, rin_b, rout_b, act_b1, act_b2, gate_b1, gate_b2,
      rtW, rinW, routW, tp00W, tp01W, tp10W, tp11W, act1W, act2W, g1W, g2W,
      sum_s, sum_sg, sum_v, sum_vg);

  node_kernel<<<N / 16, 32, 0, stream>>>(
      h, sum_s, sum_sg, sum_v, sum_vg, vrW, vr_b, swsW, self_bs, swvW, out);
}
